// DecoderRNN_6975026889175
// MI455X (gfx1250) — matrix operand order repaired and checked by reading, not compiled, not run on hardware
//
#include <hip/hip_runtime.h>
#include <math.h>

// ---------------- dims ----------------
#define Bv   64
#define Nv   49
#define ENCv 2048
#define Sv   26
#define Vv   10000
#define Ev   512
#define Hv   512
#define Av   512

typedef __attribute__((ext_vector_type(16))) _Float16 v16h;
typedef __attribute__((ext_vector_type(8)))  float    v8f;

union V16 { v16h v; _Float16 h[16]; };

__device__ __forceinline__ void cvt8(const float4& u, const float4& w, V16& dst, int base) {
    dst.h[base + 0] = (_Float16)u.x; dst.h[base + 1] = (_Float16)u.y;
    dst.h[base + 2] = (_Float16)u.z; dst.h[base + 3] = (_Float16)u.w;
    dst.h[base + 4] = (_Float16)w.x; dst.h[base + 5] = (_Float16)w.y;
    dst.h[base + 6] = (_Float16)w.z; dst.h[base + 7] = (_Float16)w.w;
}

// Partial GEMM: each wave owns (M-tile, N-group of NT tiles, K-chunk ks).
// DIRECT: out[m*ldc+n] = acc + bias[n]   (requires nks==1)
// else  : out[ks*M*N + m*N + n] = acc    (partial slab, reduced later)
// fp32 in, f16 WMMA f32-accum. M%16==0, (N/16)%NT==0, kchunk%32==0.
template <int NT, bool DIRECT>
__global__ void k_gemm(const float* __restrict__ A, int lda,
                       const float* __restrict__ W, int ldw,
                       const float* __restrict__ bias,
                       float* __restrict__ out, long long ldc,
                       int M, int N, int K, int kchunk, int nks)
{
    int wid    = blockIdx.x * (blockDim.x >> 5) + (threadIdx.x >> 5);
    int lane   = threadIdx.x & 31;
    int tilesN = N >> 4, tilesM = M >> 4;
    int groupsN = tilesN / NT;
    if (wid >= tilesM * groupsN * nks) return;   // whole-wave guard: EXEC all-1s
    int ks  = wid % nks;
    int rem = wid / nks;
    int gn  = rem % groupsN, tm = rem / groupsN;
    int tn0 = gn * NT;
    int kbeg = ks * kchunk;
    int kend = kbeg + kchunk; if (kend > K) kend = K;

    int half  = lane >> 4;                       // 0 or 1
    int idx16 = lane & 15;

    const float* Arow = A + (size_t)(tm * 16 + idx16) * lda;     // A row m = idx16
    const float* Wrow[NT];
#pragma unroll
    for (int j = 0; j < NT; ++j)
        Wrow[j] = W + (size_t)((tn0 + j) * 16 + idx16) * ldw;    // B column n = idx16

    v8f acc[NT] = {};
    for (int k0 = kbeg; k0 < kend; k0 += 32) {
        // ---- issue ALL raw loads first (independent dests -> clustered loads,
        //      progressive s_wait_loadcnt instead of load->wait-0 lockstep) ----
        const float4* Ap0 = (const float4*)(Arow + k0 + half * 8);
        const float4* Ap1 = (const float4*)(Arow + k0 + 16 + half * 8);
        float4 ar0 = Ap0[0], ar1 = Ap0[1], ar2 = Ap1[0], ar3 = Ap1[1];
        float4 br[NT][4];
#pragma unroll
        for (int j = 0; j < NT; ++j) {
            const float4* p = (const float4*)(Wrow[j] + k0 + half * 8);
            const float4* q = (const float4*)(Wrow[j] + k0 + 16 + half * 8);
            br[j][0] = p[0]; br[j][1] = p[1]; br[j][2] = q[0]; br[j][3] = q[1];
        }
        // ---- convert to f16 fragments ----
        // A 16x32: elems 0..7 -> K=k0+half*8+0..7 ; elems 8..15 -> K=k0+16+half*8+0..7
        V16 a;
        cvt8(ar0, ar1, a, 0);
        cvt8(ar2, ar3, a, 8);
        // B 32x16: elems 0..7 -> K=k0+half*8+0..7 ; elems 8..15 -> K=k0+16+half*8+0..7
        V16 b[NT];
#pragma unroll
        for (int j = 0; j < NT; ++j) { cvt8(br[j][0], br[j][1], b[j], 0); cvt8(br[j][2], br[j][3], b[j], 8); }
        // ---- NT independent WMMAs (no accumulator RAW between them) ----
#pragma unroll
        for (int j = 0; j < NT; ++j)
            acc[j] = __builtin_amdgcn_wmma_f32_16x16x32_f16(false, a.v, false, b[j].v,
                                                            (short)0, acc[j], false, false);
    }
    // C/D layout: VGPR r -> M = r + half*8 (within tile), N = idx16
#pragma unroll
    for (int j = 0; j < NT; ++j) {
        int ncol = tn0 * 16 + j * 16 + idx16;
        if (DIRECT) {
            float bv = bias[ncol];
            float* Cb = out + (size_t)(tm * 16 + half * 8) * (size_t)ldc + (size_t)ncol;
#pragma unroll
            for (int r = 0; r < 8; ++r) Cb[(size_t)r * (size_t)ldc] = acc[j][r] + bv;
        } else {
            float* Pb = out + (size_t)ks * (size_t)M * (size_t)N
                            + (size_t)(tm * 16 + half * 8) * (size_t)N + (size_t)ncol;
#pragma unroll
            for (int r = 0; r < 8; ++r) Pb[(size_t)r * (size_t)N] = acc[j][r];
        }
    }
}

// C[m*ldc+n] = sum_ks part[ks*M*N + m*N + n] + bias[n] (+ bias2[n])
__global__ void k_reduce(const float* __restrict__ part, int nslabs,
                         const float* __restrict__ bias, const float* __restrict__ bias2,
                         float* __restrict__ C, long long ldc, int M, int N)
{
    int idx = blockIdx.x * 256 + threadIdx.x;
    if (idx >= M * N) return;
    int m = idx / N, n = idx % N;
    float s = bias[n];
    if (bias2) s += bias2[n];
    for (int k = 0; k < nslabs; ++k) s += part[(size_t)k * M * N + idx];
    C[(size_t)m * (size_t)ldc + n] = s;
}

// mean over N feature pixels
__global__ void k_mean(const float* __restrict__ features, float* __restrict__ mean_enc)
{
    int b = blockIdx.x, tid = threadIdx.x;
    const float* fb = features + (size_t)b * Nv * ENCv;
    for (int col = tid; col < ENCv; col += 256) {
        float s = 0.f;
        for (int n = 0; n < Nv; ++n) s += fb[(size_t)n * ENCv + col];
        mean_enc[(size_t)b * ENCv + col] = s * (1.0f / (float)Nv);
    }
}

// attention: e = relu(att1 + att2) . W_full + b_full ; softmax over N ; context = sum alpha*features
__global__ void k_attention(const float* __restrict__ att1,      // [B,N,A]
                            const float* __restrict__ att2,      // [B,A]
                            const float* __restrict__ features,  // [B,N,ENC]
                            const float* __restrict__ W_full,    // [A]
                            const float* __restrict__ b_full,    // [1]
                            float* __restrict__ context,         // [B,ENC]
                            float* __restrict__ alphas)          // offset t*N, stride (S-1)*N per b
{
    __shared__ float e_s[64];
    __shared__ float inv_s;
    int b = blockIdx.x, tid = threadIdx.x;
    int wv = tid >> 5, lane = tid & 31;

    const float* a2 = att2 + (size_t)b * Av;
    for (int n = wv; n < Nv; n += 8) {
        const float* a1 = att1 + ((size_t)b * Nv + n) * Av;
        float s = 0.f;
        for (int a = lane; a < Av; a += 32) {
            float t = a1[a] + a2[a];
            t = t > 0.f ? t : 0.f;
            s += t * W_full[a];
        }
        for (int off = 16; off > 0; off >>= 1) s += __shfl_xor(s, off, 32);
        if (lane == 0) e_s[n] = s + b_full[0];
    }
    __syncthreads();
    if (tid == 0) {
        float m = e_s[0];
        for (int n = 1; n < Nv; ++n) m = fmaxf(m, e_s[n]);
        float ssum = 0.f;
        for (int n = 0; n < Nv; ++n) { float ex = expf(e_s[n] - m); e_s[n] = ex; ssum += ex; }
        inv_s = 1.0f / ssum;
    }
    __syncthreads();
    if (tid < Nv) {
        float al = e_s[tid] * inv_s;
        e_s[tid] = al;
        alphas[(size_t)b * ((Sv - 1) * Nv) + tid] = al;
    }
    __syncthreads();
    const float* fb = features + (size_t)b * Nv * ENCv;
    for (int col = tid; col < ENCv; col += 256) {
        float s = 0.f;
        for (int n = 0; n < Nv; ++n) s += fb[(size_t)n * ENCv + col] * e_s[n];
        context[(size_t)b * ENCv + col] = s;
    }
}

// x[b] = concat(emb[captions[b,t]], context[b])
__global__ void k_build_x(const int* __restrict__ captions, int t,
                          const float* __restrict__ emb,
                          const float* __restrict__ context,
                          float* __restrict__ x)
{
    int b = blockIdx.x, tid = threadIdx.x;
    int tok = captions[b * Sv + t];
    const float* erow = emb + (size_t)tok * Ev;
    float* xr = x + (size_t)b * (Ev + ENCv);
    for (int j = tid; j < Ev; j += 256) xr[j] = erow[j];
    for (int j = tid; j < ENCv; j += 256) xr[Ev + j] = context[(size_t)b * ENCv + j];
}

// LSTM pointwise: gate order i, f, g, o
__global__ void k_lstm(const float* __restrict__ gates,  // [B, 4H]
                       float* __restrict__ h, float* __restrict__ c)
{
    int idx = blockIdx.x * 256 + threadIdx.x;
    if (idx >= Bv * Hv) return;
    int b = idx >> 9, j = idx & (Hv - 1);
    const float* g = gates + (size_t)b * 4 * Hv;
    float ig = g[j], fg = g[Hv + j], gg = g[2 * Hv + j], og = g[3 * Hv + j];
    float si = 1.f / (1.f + expf(-ig));
    float sf = 1.f / (1.f + expf(-fg));
    float so = 1.f / (1.f + expf(-og));
    float cn = sf * c[idx] + si * tanhf(gg);
    float hn = so * tanhf(cn);
    c[idx] = cn;
    h[idx] = hn;
}

template <int NT, bool DIRECT>
static inline void launch_gemm(const float* A, int lda, const float* W, int ldw,
                               const float* bias, float* out, long long ldc,
                               int M, int N, int K, int kchunk, int nks, hipStream_t s)
{
    int waves  = (M / 16) * ((N / 16) / NT) * nks;
    int blocks = (waves + 7) / 8;
    k_gemm<NT, DIRECT><<<blocks, 256, 0, s>>>(A, lda, W, ldw, bias, out, ldc, M, N, K, kchunk, nks);
}

extern "C" void kernel_launch(void* const* d_in, const int* in_sizes, int n_in,
                              void* d_out, int out_size, void* d_ws, size_t ws_size,
                              hipStream_t stream)
{
    const float* features  = (const float*)d_in[0];
    const int*   captions  = (const int*)  d_in[1];
    const float* emb       = (const float*)d_in[2];
    const float* W_ih      = (const float*)d_in[3];
    const float* b_ih      = (const float*)d_in[4];
    const float* W_hh      = (const float*)d_in[5];
    const float* b_hh      = (const float*)d_in[6];
    const float* W_enc_att = (const float*)d_in[7];
    const float* b_enc_att = (const float*)d_in[8];
    const float* W_dec_att = (const float*)d_in[9];
    const float* b_dec_att = (const float*)d_in[10];
    const float* W_full    = (const float*)d_in[11];
    const float* b_full    = (const float*)d_in[12];
    const float* W_init_h  = (const float*)d_in[13];
    const float* b_init_h  = (const float*)d_in[14];
    const float* W_init_c  = (const float*)d_in[15];
    const float* b_init_c  = (const float*)d_in[16];
    const float* W_fcn     = (const float*)d_in[17];
    const float* b_fcn     = (const float*)d_in[18];

    float* preds  = (float*)d_out;                                   // [B, S-1, V]
    float* alphas = preds + (size_t)Bv * (Sv - 1) * Vv;              // [B, S-1, N]

    // workspace layout (fp32 elements)
    float* ws       = (float*)d_ws;
    float* mean_enc = ws;                           // 64*2048
    float* h        = mean_enc + Bv * ENCv;         // 64*512
    float* c        = h + Bv * Hv;                  // 64*512
    float* att2     = c + Bv * Hv;                  // 64*512
    float* context  = att2 + Bv * Av;               // 64*2048
    float* x        = context + Bv * ENCv;          // 64*2560
    float* gates    = x + Bv * (Ev + ENCv);         // 64*2048
    float* att1     = gates + Bv * 4 * Hv;          // 3136*512
    float* part_g   = att1 + (size_t)Bv * Nv * Av;  // 5 * 64*2048  (4 ih slabs + 1 hh slab)
    float* part_p   = part_g + 5 * Bv * 4 * Hv;     // 2 * 64*10000
    float* part_a   = part_p + 2 * Bv * Vv;         // 2 * 64*512

    // ---- one-time phase ----
    k_mean<<<Bv, 256, 0, stream>>>(features, mean_enc);
    launch_gemm<2, true>(mean_enc, ENCv, W_init_h, ENCv, b_init_h, h, Hv, Bv, Hv, ENCv, ENCv, 1, stream);
    launch_gemm<2, true>(mean_enc, ENCv, W_init_c, ENCv, b_init_c, c, Hv, Bv, Hv, ENCv, ENCv, 1, stream);
    // att1 = features @ W_enc_att^T + b  (view features as [B*N, ENC]) : 3136 waves
    launch_gemm<2, true>(features, ENCv, W_enc_att, ENCv, b_enc_att, att1, Av, Bv * Nv, Av, ENCv, ENCv, 1, stream);

    // ---- 25 sequential decode steps ----
    for (int t = 0; t < Sv - 1; ++t) {
        // att2 = h @ W_dec_att^T + b   (K split x2 -> 128 waves, then reduce)
        launch_gemm<2, false>(h, Hv, W_dec_att, Hv, nullptr, part_a, 0, Bv, Av, Hv, 256, 2, stream);
        k_reduce<<<(Bv * Av + 255) / 256, 256, 0, stream>>>(part_a, 2, b_dec_att, nullptr,
                                                            att2, Av, Bv, Av);
        // attention: softmax weights + context; alphas -> d_out
        k_attention<<<Bv, 256, 0, stream>>>(att1, att2, features, W_full, b_full,
                                            context, alphas + (size_t)t * Nv);
        // x = [embed_t, context]
        k_build_x<<<Bv, 256, 0, stream>>>(captions, t, emb, context, x);
        // gates = x @ W_ih^T + h @ W_hh^T + b_ih + b_hh
        //   ih: K=2560 split x4 (kchunk 640) -> 1024 waves, slabs 0..3
        launch_gemm<2, false>(x, Ev + ENCv, W_ih, Ev + ENCv, nullptr, part_g, 0,
                              Bv, 4 * Hv, Ev + ENCv, 640, 4, stream);
        //   hh: K=512 -> 256 waves, slab 4
        launch_gemm<2, false>(h, Hv, W_hh, Hv, nullptr, part_g + (size_t)4 * Bv * 4 * Hv, 0,
                              Bv, 4 * Hv, Hv, 512, 1, stream);
        k_reduce<<<(Bv * 4 * Hv + 255) / 256, 256, 0, stream>>>(part_g, 5, b_ih, b_hh,
                                                                gates, 4 * Hv, Bv, 4 * Hv);
        // LSTM pointwise (updates h, c in place)
        k_lstm<<<(Bv * Hv + 255) / 256, 256, 0, stream>>>(gates, h, c);
        // pred_t = h_new @ W_fcn^T + b_fcn : NT=1 (625 tiles), K split x2 -> 5000 waves
        launch_gemm<1, false>(h, Hv, W_fcn, Hv, nullptr, part_p, 0, Bv, Vv, Hv, 256, 2, stream);
        k_reduce<<<(Bv * Vv + 255) / 256, 256, 0, stream>>>(part_p, 2, b_fcn, nullptr,
                                                            preds + (size_t)t * Vv,
                                                            (long long)(Sv - 1) * Vv, Bv, Vv);
    }
}
